// FeatureEnhancerLayer_19842748907852
// MI455X (gfx1250) — compile-verified, matrix operand order checked by reading, not run on hardware
//
#include <hip/hip_runtime.h>
#include <hip/hip_fp16.h>
#include <math.h>
#include <stdint.h>

// ---------------------------------------------------------------------------
// MI455X / gfx1250 implementation. wave32. All GEMM-shaped math goes through
// v_wmma_f32_16x16x32_f16 (f16 in, f32 accum). GEMM tiles are staged with the
// async global->LDS engine (ASYNCcnt) and double-buffered; attention K tiles
// are staged with the Tensor Data Mover (TENSORcnt).
// ---------------------------------------------------------------------------

typedef _Float16 v16h __attribute__((ext_vector_type(16)));
typedef float    v8f  __attribute__((ext_vector_type(8)));
typedef unsigned v4u  __attribute__((ext_vector_type(4)));
typedef unsigned v8u  __attribute__((ext_vector_type(8)));

#define WMMA16(a, b, c) \
  __builtin_amdgcn_wmma_f32_16x16x32_f16(false, (a), false, (b), (short)0, (c), false, false)

#define NHEADS 8
#define DHEAD  32
#define NLVL   4
#define NPTS   4

// Async global->LDS 16B copy: VDST carries the per-lane LDS byte address
// (low 32 bits of the generic pointer == DS-space offset), VADDR the 64-bit
// global address. Tracked by ASYNCcnt.
__device__ __forceinline__ void g2l_async_b128(const _Float16* g, _Float16* l) {
  unsigned lo = (unsigned)(uintptr_t)l;
  unsigned long long ga = (unsigned long long)(uintptr_t)g;
  asm volatile("global_load_async_to_lds_b128 %0, %1, off"
               :: "v"(lo), "v"(ga) : "memory");
}
__device__ __forceinline__ void async_wait0() {
  asm volatile("s_wait_asynccnt 0" ::: "memory");
}

// ---------------------------------------------------------------------------
// Generic WMMA GEMM: C[M,N] = A[M,K] @ B[N,K]^T + bias[N], optional ReLU,
// optional f32 and/or f16 outputs (same layout, ldc).
// Block: 256 threads = 8 waves; block tile 64(M) x 128(N); wave tile 32x32.
// Double-buffered async LDS staging: next k-tile DMA overlaps current WMMAs.
// ---------------------------------------------------------------------------
__launch_bounds__(256)
__global__ void fel_gemm_wmma(const _Float16* __restrict__ A,
                              const _Float16* __restrict__ B,
                              const float* __restrict__ bias,
                              float* __restrict__ Cf,
                              _Float16* __restrict__ Ch,
                              int M, int N, int K, int ldc, int relu) {
  __shared__ _Float16 As[2][64 * 32];
  __shared__ _Float16 Bs[2][128 * 32];

  const int tid  = threadIdx.x;
  const int wave = tid >> 5;
  const int lane = tid & 31;
  const int m0 = blockIdx.y * 64;
  const int n0 = blockIdx.x * 128;
  const int waveM = (wave >> 2) * 32;   // 0 or 32
  const int waveN = (wave & 3) * 32;    // 0..96
  const int fr = lane & 15;
  const int fk = (lane >> 4) * 16;

  // per-thread staging coordinates
  const int ar = tid >> 2, ac = (tid & 3) * 8;    // A: 64 x 32, 8 halves/thread
  const int br = tid >> 1, bc = (tid & 1) * 16;   // B: 128 x 32, 16 halves/thread
  int gra = m0 + ar; if (gra >= M) gra = M - 1;
  const _Float16* Abase = A + (size_t)gra * K + ac;
  const _Float16* Bbase = B + (size_t)(n0 + br) * K + bc;

  auto stage = [&](int k, int buf) {
    g2l_async_b128(Abase + k, &As[buf][ar * 32 + ac]);
    g2l_async_b128(Bbase + k, &Bs[buf][br * 32 + bc]);
    g2l_async_b128(Bbase + k + 8, &Bs[buf][br * 32 + bc + 8]);
  };

  v8f acc00 = {}, acc01 = {}, acc10 = {}, acc11 = {};

  stage(0, 0);
  async_wait0();
  __syncthreads();

  int buf = 0;
  for (int k = 0; k < K; k += 32) {
    if (k + 32 < K) stage(k + 32, buf ^ 1);   // overlaps with WMMAs below

    v16h a0 = *(const v16h*)(&As[buf][(waveM +      fr) * 32 + fk]);
    v16h a1 = *(const v16h*)(&As[buf][(waveM + 16 + fr) * 32 + fk]);
    v16h b0 = *(const v16h*)(&Bs[buf][(waveN +      fr) * 32 + fk]);
    v16h b1 = *(const v16h*)(&Bs[buf][(waveN + 16 + fr) * 32 + fk]);
    acc00 = WMMA16(a0, b0, acc00);
    acc01 = WMMA16(a0, b1, acc01);
    acc10 = WMMA16(a1, b0, acc10);
    acc11 = WMMA16(a1, b1, acc11);

    async_wait0();
    __syncthreads();
    buf ^= 1;
  }

  // Store: C-fragment layout — lanes 0-15 hold rows j, lanes 16-31 rows j+8,
  // column = waveN + tn*16 + (lane&15).
  const int rsel = (lane >> 4) * 8;
  v8f accs[2][2] = {{acc00, acc01}, {acc10, acc11}};
  for (int tm = 0; tm < 2; ++tm) {
    for (int tn = 0; tn < 2; ++tn) {
      int col = n0 + waveN + tn * 16 + fr;
      float bv = bias ? bias[col] : 0.0f;
      for (int j = 0; j < 8; ++j) {
        int row = m0 + waveM + tm * 16 + j + rsel;
        if (row < M) {
          float v = accs[tm][tn][j] + bv;
          if (relu) v = fmaxf(v, 0.0f);
          if (Cf) Cf[(size_t)row * ldc + col] = v;
          if (Ch) Ch[(size_t)row * ldc + col] = (_Float16)v;
        }
      }
    }
  }
}

// ---------------------------------------------------------------------------
// WMMA flash attention (streaming exp-sum softmax, dh = 32 = WMMA K depth).
// grid.y = b*NH + h; grid.x tiles 128 q-rows; each wave owns a 16-row q tile.
// K chunks (64 x 32 f16, strided rows) are fetched by the Tensor Data Mover;
// V chunks are transposed through registers into LDS.
// ---------------------------------------------------------------------------
__launch_bounds__(256)
__global__ void fel_attn_wmma(const _Float16* __restrict__ Q, int qStride,
                              const _Float16* __restrict__ Kv,
                              const _Float16* __restrict__ Vv, int kvStride,
                              float* __restrict__ O,
                              int Lq, int Lk, float scale) {
  __shared__ _Float16 Ks[64 * 32];       // K chunk, [key][dh]  (TDM-filled)
  __shared__ _Float16 VTs[32 * 64];      // V chunk transposed, [dh][key]
  __shared__ _Float16 Ps[8][16 * 32];    // per-wave P tile (16 q x 32 keys)

  const int bh = blockIdx.y;
  const int b = bh >> 3, h = bh & 7;
  const int tid = threadIdx.x, wave = tid >> 5, lane = tid & 31;
  const int fr = lane & 15;
  const int fk = (lane >> 4) * 16;
  const int rsel = (lane >> 4) * 8;
  const int q0 = blockIdx.x * 128 + wave * 16;

  int qr = q0 + fr; if (qr >= Lq) qr = Lq - 1;
  const _Float16* qrow = Q + ((size_t)b * Lq + qr) * qStride + h * DHEAD;
  v16h qf = *(const v16h*)(qrow + fk);

  const _Float16* kb = Kv + (size_t)b * Lk * kvStride + h * DHEAD;
  const _Float16* vb = Vv + (size_t)b * Lk * kvStride + h * DHEAD;
  const unsigned ldsK = (unsigned)(uintptr_t)(&Ks[0]);

  v8f num0 = {}, num1 = {}, den = {};
  _Float16* P = &Ps[wave][0];

  for (int kc = 0; kc < Lk; kc += 64) {
    __syncthreads();
    if (wave == 0) {
      // Tensor DMA descriptor: 2D tile, 32 halves x 64 rows, 2B elements,
      // row stride kvStride; tensor_dim1 = remaining rows => OOB rows zero.
      unsigned long long ga = (unsigned long long)(uintptr_t)(kb + (size_t)kc * kvStride);
      unsigned rem = (unsigned)(Lk - kc);
      v4u g0;
      g0[0] = 1u;                                        // count = 1 valid D#
      g0[1] = ldsK;                                      // lds_addr
      g0[2] = (unsigned)ga;                              // global_addr[31:0]
      g0[3] = (unsigned)((ga >> 32) & 0x1FFFFFFu) | (2u << 30);  // addr[56:32], type=2
      v8u g1;
      g1[0] = 0x00010000u;                               // data_size = 1 (2 bytes)
      g1[1] = 32u << 16;                                 // tensor_dim0 = 32
      g1[2] = (rem & 0xFFFFu) << 16;                     // tensor_dim1 lo
      g1[3] = ((rem >> 16) & 0xFFFFu) | (32u << 16);     // tensor_dim1 hi | tile_dim0 = 32
      g1[4] = 64u;                                       // tile_dim1 = 64 rows
      g1[5] = (unsigned)kvStride;                        // tensor_dim0_stride lo
      g1[6] = 0u;                                        // stride hi | dim1_stride lo
      g1[7] = 0u;
      asm volatile("tensor_load_to_lds %0, %1" :: "s"(g0), "s"(g1) : "memory");
      __builtin_amdgcn_s_wait_tensorcnt(0);
    }
    {  // V chunk: register transpose into VTs
      int r = tid >> 2;            // key 0..63
      int c = (tid & 3) * 8;       // dh offset 0,8,16,24
      int gk = kc + r; if (gk >= Lk) gk = Lk - 1;
      const _Float16* vr = vb + (size_t)gk * kvStride + c;
      if (kc + 64 < Lk) __builtin_prefetch(vr + (size_t)64 * kvStride, 0, 1);
      for (int i = 0; i < 8; ++i) VTs[(c + i) * 64 + r] = vr[i];
    }
    __syncthreads();

    for (int half = 0; half < 2; ++half) {
      const int kb0 = half * 32;
      // scores: two 16-key WMMAs (K depth = dh = 32)
      v16h kf0 = *(const v16h*)(&Ks[(kb0 +      fr) * 32 + fk]);
      v16h kf1 = *(const v16h*)(&Ks[(kb0 + 16 + fr) * 32 + fk]);
      v8f s0 = {}, s1 = {};
      s0 = WMMA16(qf, kf0, s0);
      s1 = WMMA16(qf, kf1, s1);

      const int col0 = kc + kb0 + fr;   // global key index of s0 column
      const int col1 = col0 + 16;
      for (int j = 0; j < 8; ++j) {
        float e0 = (col0 < Lk) ? __expf(s0[j] * scale) : 0.0f;
        float e1 = (col1 < Lk) ? __expf(s1[j] * scale) : 0.0f;
        int pr = j + rsel;
        P[pr * 32 + fr]      = (_Float16)e0;
        P[pr * 32 + 16 + fr] = (_Float16)e1;
        // row-sum: reduce over the 16 lanes holding this row's columns
        float rs = e0 + e1;
        rs += __shfl_xor(rs, 1, 16);
        rs += __shfl_xor(rs, 2, 16);
        rs += __shfl_xor(rs, 4, 16);
        rs += __shfl_xor(rs, 8, 16);
        den[j] += rs;
      }
      // P @ V : A = P(16x32 keys), B = V(32 keys x 16 dh) per dh half
      v16h pf  = *(const v16h*)(&P[fr * 32 + fk]);
      v16h vf0 = *(const v16h*)(&VTs[(     fr) * 64 + kb0 + fk]);
      v16h vf1 = *(const v16h*)(&VTs[(16 + fr) * 64 + kb0 + fk]);
      num0 = WMMA16(pf, vf0, num0);
      num1 = WMMA16(pf, vf1, num1);
    }
  }

  for (int j = 0; j < 8; ++j) {
    int row = q0 + j + rsel;
    if (row < Lq) {
      float d = den[j];
      float inv = (d > 0.0f) ? 1.0f / d : 0.0f;
      float* orow = O + ((size_t)b * Lq + row) * 256 + h * DHEAD;
      orow[fr]      = num0[j] * inv;
      orow[16 + fr] = num1[j] * inv;
    }
  }
}

// ---------------------------------------------------------------------------
// Deformable sampling: one thread per (b, q, h). Fused 16-way softmax over
// attention logits + 4 levels x 4 points bilinear gather from the f32 value
// tensor (L2-resident). acc over the 32-wide head channel.
// ---------------------------------------------------------------------------
__launch_bounds__(256)
__global__ void fel_deform_sample(const float* __restrict__ V,        // [b, LV, 256]
                                  const float* __restrict__ attnlog,  // [b, NQ, 128]
                                  const float* __restrict__ offb,     // [b, NQ, 256]
                                  const float* __restrict__ ref,      // [b, NQ, NL, 2]
                                  const int*   __restrict__ shapes,   // [NL][2] (H,W)
                                  float* __restrict__ out,            // [b, NQ, 256]
                                  int NQ, int LV) {
  int idx = blockIdx.x * blockDim.x + threadIdx.x;
  int total = 2 * NQ * NHEADS;
  if (idx >= total) return;
  int h = idx & 7;
  int t = idx >> 3;
  int q = t % NQ;
  int b = t / NQ;

  // softmax over NL*NP = 16 logits for this head
  const float* lg = attnlog + (size_t)(b * NQ + q) * 128 + h * 16;
  float w[16], mx = -1e30f;
  for (int i = 0; i < 16; ++i) { w[i] = lg[i]; mx = fmaxf(mx, w[i]); }
  float s = 0.0f;
  for (int i = 0; i < 16; ++i) { w[i] = __expf(w[i] - mx); s += w[i]; }
  float invs = 1.0f / s;

  float acc[DHEAD];
  for (int d = 0; d < DHEAD; ++d) acc[d] = 0.0f;

  const float* offr = offb + (size_t)(b * NQ + q) * 256 + h * 32;
  const float* refr = ref + (size_t)(b * NQ + q) * NLVL * 2;

  int start = 0;
  for (int l = 0; l < NLVL; ++l) {
    int H = shapes[l * 2], W = shapes[l * 2 + 1];
    float rx = refr[l * 2], ry = refr[l * 2 + 1];
    for (int p = 0; p < NPTS; ++p) {
      float ox = offr[l * 8 + p * 2], oy = offr[l * 8 + p * 2 + 1];
      float x = (rx + ox / (float)W) * W - 0.5f;
      float y = (ry + oy / (float)H) * H - 0.5f;
      float x0 = floorf(x), y0 = floorf(y);
      float wx = x - x0, wy = y - y0;
      int xi = (int)x0, yi = (int)y0;
      float aw = w[l * 4 + p] * invs;
      for (int ty = 0; ty < 2; ++ty) {
        for (int tx = 0; tx < 2; ++tx) {
          int yy = yi + ty, xx = xi + tx;
          if (yy < 0 || yy >= H || xx < 0 || xx >= W) continue;
          float wgt = aw * (ty ? wy : 1.0f - wy) * (tx ? wx : 1.0f - wx);
          const float4* v4 =
              (const float4*)(V + ((size_t)b * LV + start + yy * W + xx) * 256 + h * 32);
          for (int d = 0; d < 8; ++d) {
            float4 vv = v4[d];
            acc[d * 4 + 0] += wgt * vv.x;
            acc[d * 4 + 1] += wgt * vv.y;
            acc[d * 4 + 2] += wgt * vv.z;
            acc[d * 4 + 3] += wgt * vv.w;
          }
        }
      }
    }
    start += H * W;
  }
  float* orow = out + (size_t)(b * NQ + q) * 256 + h * 32;
  for (int d = 0; d < DHEAD; ++d) orow[d] = acc[d];
}

// ---------------------------------------------------------------------------
// glue: f32 -> f16 convert; (a+b) -> f16
// ---------------------------------------------------------------------------
__global__ void fel_cvt_h(const float* __restrict__ a, _Float16* __restrict__ o, size_t n) {
  size_t i = (size_t)blockIdx.x * blockDim.x + threadIdx.x;
  if (i < n) o[i] = (_Float16)a[i];
}
__global__ void fel_add_cvt_h(const float* __restrict__ a, const float* __restrict__ b,
                              _Float16* __restrict__ o, size_t n) {
  size_t i = (size_t)blockIdx.x * blockDim.x + threadIdx.x;
  if (i < n) o[i] = (_Float16)(a[i] + b[i]);
}

// ---------------------------------------------------------------------------
extern "C" void kernel_launch(void* const* d_in, const int* in_sizes, int n_in,
                              void* d_out, int out_size, void* d_ws, size_t ws_size,
                              hipStream_t stream) {
  const int BS = 2, NQ = 13294, LT = 256, D = 256;
  const int MI = BS * NQ;   // 26588 image rows
  const int MT = BS * LT;   // 512 text rows
  const float scale = 0.17677669529663687f;  // 1/sqrt(32)

  const float* img    = (const float*)d_in[0];
  const float* txt    = (const float*)d_in[1];
  const float* refpts = (const float*)d_in[2];
  const float* W_val  = (const float*)d_in[3];
  const float* b_val  = (const float*)d_in[4];
  const float* W_off  = (const float*)d_in[5];
  const float* b_off  = (const float*)d_in[6];
  const float* W_attn = (const float*)d_in[7];
  const float* b_attn = (const float*)d_in[8];
  const float* W_dout = (const float*)d_in[9];
  const float* b_dout = (const float*)d_in[10];
  const float* Wqkv_t   = (const float*)d_in[11];
  const float* bqkv_t   = (const float*)d_in[12];
  const float* Wo_t     = (const float*)d_in[13];
  const float* bo_t     = (const float*)d_in[14];
  const float* Wqkv_i2t = (const float*)d_in[15];
  const float* bqkv_i2t = (const float*)d_in[16];
  const float* Wo_i2t   = (const float*)d_in[17];
  const float* bo_i2t   = (const float*)d_in[18];
  const float* Wqkv_t2i = (const float*)d_in[19];
  const float* bqkv_t2i = (const float*)d_in[20];
  const float* Wo_t2i   = (const float*)d_in[21];
  const float* bo_t2i   = (const float*)d_in[22];
  const float* W1 = (const float*)d_in[23];
  const float* b1 = (const float*)d_in[24];
  const float* W2 = (const float*)d_in[25];
  const float* b2 = (const float*)d_in[26];
  const int* shapes = (const int*)d_in[27];
  float* out = (float*)d_out;

  // -------- workspace bump allocator --------
  char* base = (char*)d_ws;
  size_t off = 0;
  auto aF32 = [&](size_t n) -> float* {
    float* p = (float*)(base + off);
    off = (off + n * 4 + 255) & ~(size_t)255;
    return p;
  };
  auto aF16 = [&](size_t n) -> _Float16* {
    _Float16* p = (_Float16*)(base + off);
    off = (off + n * 2 + 255) & ~(size_t)255;
    return p;
  };

  // f16 activation copies + weights
  _Float16* img16 = aF16((size_t)MI * D);
  _Float16* txt16 = aF16((size_t)MT * D);
  _Float16* wval16   = aF16(256 * 256);
  _Float16* woff16   = aF16(256 * 256);
  _Float16* wattn16  = aF16(128 * 256);
  _Float16* wdout16  = aF16(256 * 256);
  _Float16* wqkvt16  = aF16(768 * 256);
  _Float16* wot16    = aF16(256 * 256);
  _Float16* wqkvi16  = aF16(768 * 256);
  _Float16* woi16    = aF16(256 * 256);
  _Float16* wqkvti16 = aF16(768 * 256);
  _Float16* woti16   = aF16(256 * 256);
  _Float16* w1_16 = aF16(1024 * 256);
  _Float16* w2_16 = aF16(256 * 1024);

  float* vval    = aF32((size_t)MI * 256);
  float* offb    = aF32((size_t)MI * 256);
  float* attnlog = aF32((size_t)MI * 128);
  float* samp    = aF32((size_t)MI * 256);
  float* imgAttn = aF32((size_t)MI * 256);
  float* otxt    = aF32((size_t)MT * 256);
  float* txtSA   = aF32((size_t)MT * 256);
  float* oi2t    = aF32((size_t)MT * 256);
  float* i2tout  = aF32((size_t)MT * 256);
  float* ot2i    = aF32((size_t)MI * 256);
  float* t2iout  = aF32((size_t)MI * 256);

  _Float16* samp16    = aF16((size_t)MI * 256);
  _Float16* imgAttn16 = aF16((size_t)MI * 256);
  _Float16* qkvt16    = aF16((size_t)MT * 768);
  _Float16* otxt16    = aF16((size_t)MT * 256);
  _Float16* txtSA16   = aF16((size_t)MT * 256);
  _Float16* qi2t16    = aF16((size_t)MT * 256);
  _Float16* kvi2t16   = aF16((size_t)MI * 512);
  _Float16* oi2t16    = aF16((size_t)MT * 256);
  _Float16* qt2i16    = aF16((size_t)MI * 256);
  _Float16* kvt2i16   = aF16((size_t)MT * 512);
  _Float16* ot2i16    = aF16((size_t)MI * 256);
  _Float16* himg16    = aF16((size_t)MI * 256);
  _Float16* ffn1i16   = aF16((size_t)MI * 1024);
  _Float16* htxt16    = aF16((size_t)MT * 256);
  _Float16* ffn1t16   = aF16((size_t)MT * 1024);
  (void)ws_size; (void)in_sizes; (void)n_in; (void)out_size;

  auto cvt = [&](const float* a, _Float16* o, size_t n) {
    fel_cvt_h<<<dim3((unsigned)((n + 255) / 256)), 256, 0, stream>>>(a, o, n);
  };
  auto gemm = [&](const _Float16* A, const _Float16* B, const float* bias,
                  float* Cf, _Float16* Ch, int M, int N, int K, int ldc, int relu) {
    dim3 g(N / 128, (M + 63) / 64);
    fel_gemm_wmma<<<g, 256, 0, stream>>>(A, B, bias, Cf, Ch, M, N, K, ldc, relu);
  };
  auto attn = [&](const _Float16* Q, int qs, const _Float16* K, const _Float16* V,
                  int kvs, float* O, int Lq, int Lk) {
    dim3 g((Lq + 127) / 128, BS * NHEADS);
    fel_attn_wmma<<<g, 256, 0, stream>>>(Q, qs, K, V, kvs, O, Lq, Lk, scale);
  };

  // -------- f16 conversions (inputs + weights) --------
  cvt(img, img16, (size_t)MI * D);
  cvt(txt, txt16, (size_t)MT * D);
  cvt(W_val, wval16, 256 * 256);
  cvt(W_off, woff16, 256 * 256);
  cvt(W_attn, wattn16, 128 * 256);
  cvt(W_dout, wdout16, 256 * 256);
  cvt(Wqkv_t, wqkvt16, 768 * 256);
  cvt(Wo_t, wot16, 256 * 256);
  cvt(Wqkv_i2t, wqkvi16, 768 * 256);
  cvt(Wo_i2t, woi16, 256 * 256);
  cvt(Wqkv_t2i, wqkvti16, 768 * 256);
  cvt(Wo_t2i, woti16, 256 * 256);
  cvt(W1, w1_16, 1024 * 256);
  cvt(W2, w2_16, 256 * 1024);

  // -------- deformable attention on image path --------
  gemm(img16, wval16, b_val, vval, nullptr, MI, 256, 256, 256, 0);      // value proj
  gemm(img16, woff16, b_off, offb, nullptr, MI, 256, 256, 256, 0);      // offsets
  gemm(img16, wattn16, b_attn, attnlog, nullptr, MI, 128, 256, 128, 0); // attn logits
  {
    int total = BS * NQ * NHEADS;
    fel_deform_sample<<<dim3((total + 255) / 256), 256, 0, stream>>>(
        vval, attnlog, offb, refpts, shapes, samp, NQ, NQ);
  }
  cvt(samp, samp16, (size_t)MI * 256);
  gemm(samp16, wdout16, b_dout, imgAttn, imgAttn16, MI, 256, 256, 256, 0);

  // -------- text self-attention --------
  gemm(txt16, wqkvt16, bqkv_t, nullptr, qkvt16, MT, 768, 256, 768, 0);
  attn(qkvt16, 768, qkvt16 + 256, qkvt16 + 512, 768, otxt, LT, LT);
  cvt(otxt, otxt16, (size_t)MT * 256);
  gemm(otxt16, wot16, bo_t, txtSA, txtSA16, MT, 256, 256, 256, 0);

  // -------- i2t: text queries image --------
  gemm(txtSA16, wqkvi16, bqkv_i2t, nullptr, qi2t16, MT, 256, 256, 256, 0);
  gemm(imgAttn16, wqkvi16 + 256 * 256, bqkv_i2t + 256, nullptr, kvi2t16, MI, 512, 256, 512, 0);
  attn(qi2t16, 256, kvi2t16, kvi2t16 + 256, 512, oi2t, LT, NQ);
  cvt(oi2t, oi2t16, (size_t)MT * 256);
  gemm(oi2t16, woi16, bo_i2t, i2tout, nullptr, MT, 256, 256, 256, 0);

  // -------- t2i: image queries text --------
  gemm(imgAttn16, wqkvti16, bqkv_t2i, nullptr, qt2i16, MI, 256, 256, 256, 0);
  gemm(txtSA16, wqkvti16 + 256 * 256, bqkv_t2i + 256, nullptr, kvt2i16, MT, 512, 256, 512, 0);
  attn(qt2i16, 256, kvt2i16, kvt2i16 + 256, 512, ot2i, NQ, LT);
  cvt(ot2i, ot2i16, (size_t)MI * 256);
  gemm(ot2i16, woti16, bo_t2i, t2iout, nullptr, MI, 256, 256, 256, 0);

  // -------- shared FFN --------
  {
    size_t n = (size_t)MI * 256;
    fel_add_cvt_h<<<dim3((unsigned)((n + 255) / 256)), 256, 0, stream>>>(imgAttn, t2iout, himg16, n);
  }
  gemm(himg16, w1_16, b1, nullptr, ffn1i16, MI, 1024, 256, 1024, 1);
  gemm(ffn1i16, w2_16, b2, out, nullptr, MI, 256, 1024, 256, 0);
  {
    size_t n = (size_t)MT * 256;
    fel_add_cvt_h<<<dim3((unsigned)((n + 255) / 256)), 256, 0, stream>>>(txtSA, i2tout, htxt16, n);
  }
  gemm(htxt16, w1_16, b1, nullptr, ffn1t16, MT, 1024, 256, 1024, 1);
  gemm(ffn1t16, w2_16, b2, out + (size_t)MI * 256, nullptr, MT, 256, 1024, 256, 0);
}
